// SwitchFeedForward_79611513798726
// MI455X (gfx1250) — compile-verified, matrix operand order checked by reading, not run on hardware
//
#include <hip/hip_runtime.h>
#include <stdint.h>

#define T_TOK 8192
#define D_DIM 1024
#define F_DIM 4096
#define E_EXP 8
#define CAP   2048

typedef __attribute__((ext_vector_type(16))) __bf16 v16bf;
typedef __attribute__((ext_vector_type(2)))  __bf16 v2bf;
typedef __attribute__((ext_vector_type(8)))  float  v8f;
typedef __attribute__((ext_vector_type(4)))  unsigned int v4u;
typedef __attribute__((ext_vector_type(8)))  int v8i;
typedef __attribute__((ext_vector_type(4)))  int v4i;

union FragBF { v16bf v; uint4 q[2]; };

__device__ __forceinline__ unsigned fbits(float f) {
  union { float f; unsigned u; } c; c.f = f; return c.u;
}

// float -> bf16 (round-to-nearest, half-up): 2 VALU
__device__ __forceinline__ unsigned short f2bf(float f) {
  return (unsigned short)((fbits(f) + 0x8000u) >> 16);
}

// pack two floats into one dword of two bf16 (lo=a, hi=b)
__device__ __forceinline__ unsigned pack2bf(float a, float b) {
#if __has_builtin(__builtin_amdgcn_cvt_pk_bf16_f32)
  union { v2bf v; unsigned u; } c;
  c.v = __builtin_amdgcn_cvt_pk_bf16_f32(a, b);
  return c.u;
#else
  // round both, then v_perm_b32 gathers the two high halves: 3 VALU total
  unsigned au = fbits(a) + 0x8000u;
  unsigned bu = fbits(b) + 0x8000u;
  // dst bytes [0,1] = a bytes [2,3]; dst bytes [2,3] = b bytes [2,3]
  return __builtin_amdgcn_perm(bu, au, 0x07060302u);
#endif
}

// ------------------------------------------------------------- TDM A-tile load
// DMA a [128 rows x 64 cols] bf16 tile (row stride KDIM elements) into LDS.
// LDS layout after pad: row stride = 64 elems + 4 dwords pad = 72 ushorts (144B).
__device__ __forceinline__ void tdm_load_tile_a(const unsigned short* gsrc,
                                                unsigned lds_off, int kdim) {
  unsigned long long ga = (unsigned long long)(uintptr_t)gsrc;
  v4u g0;
  g0[0] = 1u;                                           // count=1 (valid user D#)
  g0[1] = lds_off;                                      // lds_addr (bytes)
  g0[2] = (unsigned)ga;                                 // global_addr[31:0]
  g0[3] = (unsigned)((ga >> 32) & 0x01FFFFFFu) | (2u << 30);  // addr[56:32] | type=2
  v8i g1 = (v8i)(0);
  const unsigned td0 = (unsigned)kdim;
  // data_size=1 (2B) | pad_enable | pad_interval=4 (32 dw) | pad_amount=3 (4 dw)
  g1[0] = (int)((1u << 16) | (1u << 20) | (4u << 22) | (3u << 25));
  g1[1] = (int)((td0 & 0xFFFFu) << 16);                 // tensor_dim0[15:0]
  g1[2] = (int)(((td0 >> 16) & 0xFFFFu) | ((unsigned)CAP << 16)); // dim0 hi | dim1 lo
  g1[3] = (int)(64u << 16);                             // tile_dim0=64 (dim1 hi=0)
  g1[4] = 128;                                          // tile_dim1=128, tile_dim2=0
  g1[5] = (int)td0;                                     // tensor_dim0_stride lo
  g1[6] = 0;                                            // stride0 hi | stride1 lo
  g1[7] = 0;
  v4i z4 = (v4i)(0);
#if __clang_major__ >= 23
  v8i z8 = (v8i)(0);
  __builtin_amdgcn_tensor_load_to_lds(g0, g1, z4, z4, z8, 0);
#else
  __builtin_amdgcn_tensor_load_to_lds(g0, g1, z4, z4, 0);
#endif
}

// ---------------------------------------------------------------- init
__global__ void init_kernel(int* __restrict__ s2t, float* __restrict__ rps) {
  int i = blockIdx.x * blockDim.x + threadIdx.x;
  if (i < E_EXP * CAP) s2t[i] = -1;
  if (i < E_EXP) rps[i] = 0.0f;
}

// ---------------------------------------------------------------- router
__global__ void router_kernel(const float* __restrict__ x,
                              const float* __restrict__ wsw,
                              const float* __restrict__ bsw,
                              int*   __restrict__ routes,
                              float* __restrict__ rpm,
                              float* __restrict__ rps) {
  const int wave = threadIdx.x >> 5;
  const int lane = threadIdx.x & 31;
  const int t = blockIdx.x * 8 + wave;
  float acc[E_EXP];
#pragma unroll
  for (int e = 0; e < E_EXP; ++e) acc[e] = 0.0f;
  const float* xp = x + (size_t)t * D_DIM;
  for (int k = lane; k < D_DIM; k += 32) {
    float xv = xp[k];
#pragma unroll
    for (int e = 0; e < E_EXP; ++e) acc[e] += xv * wsw[k * E_EXP + e];
  }
#pragma unroll
  for (int off = 16; off > 0; off >>= 1) {
#pragma unroll
    for (int e = 0; e < E_EXP; ++e) acc[e] += __shfl_xor(acc[e], off, 32);
  }
  if (lane == 0) {
    float mx = -1e30f;
#pragma unroll
    for (int e = 0; e < E_EXP; ++e) { acc[e] += bsw[e]; mx = fmaxf(mx, acc[e]); }
    float p[E_EXP]; float sum = 0.0f;
#pragma unroll
    for (int e = 0; e < E_EXP; ++e) { p[e] = __expf(acc[e] - mx); sum += p[e]; }
    float inv = 1.0f / sum;
    float pmax = -1.0f; int am = 0;
#pragma unroll
    for (int e = 0; e < E_EXP; ++e) {
      p[e] *= inv;
      if (p[e] > pmax) { pmax = p[e]; am = e; }   // first max wins
    }
    routes[t] = am;
    rpm[t] = pmax;
#pragma unroll
    for (int e = 0; e < E_EXP; ++e) atomicAdd(&rps[e], p[e]);
  }
}

// ---------------------------------------------------------------- scan
__global__ void scan_kernel(const int* __restrict__ routes,
                            int* __restrict__ pos,
                            float* __restrict__ counts_out,
                            float* __restrict__ ndrop_out) {
  const unsigned lane = threadIdx.x;
  int run[E_EXP];
#pragma unroll
  for (int e = 0; e < E_EXP; ++e) run[e] = 0;
  const unsigned below = (lane == 0) ? 0u : ((1u << lane) - 1u);
  for (int base = 0; base < T_TOK; base += 32) {
    int r = routes[base + lane];
    int p = 0;
#pragma unroll
    for (int e = 0; e < E_EXP; ++e) {
      unsigned m = (unsigned)__ballot(r == e);   // wave32: low 32 bits
      if (r == e) p = run[e] + __popc(m & below);
      run[e] += __popc(m);
    }
    pos[base + lane] = p;
  }
  if (lane < E_EXP) counts_out[lane] = (float)min(run[lane], CAP);
  if (lane == 0) {
    int tot = 0;
#pragma unroll
    for (int e = 0; e < E_EXP; ++e) tot += min(run[e], CAP);
    ndrop_out[0] = (float)(T_TOK - tot);
  }
}

// ---------------------------------------------------------------- dispatch
__global__ void dispatch_kernel(const float* __restrict__ x,
                                const int* __restrict__ routes,
                                const int* __restrict__ pos,
                                const float* __restrict__ rpm,
                                unsigned short* __restrict__ buf,
                                int* __restrict__ s2t,
                                float* __restrict__ outF) {
  const int t = blockIdx.x;
  const int r = routes[t];
  const int p = pos[t];
  const float* src = x + (size_t)t * D_DIM;
  if (p < CAP) {
    unsigned short* dst = buf + ((size_t)r * CAP + p) * D_DIM;
    for (int i = threadIdx.x * 2; i < D_DIM; i += 512) {
      float2 v = *(const float2*)&src[i];
      *(unsigned*)&dst[i] = pack2bf(v.x, v.y);
    }
    if (threadIdx.x == 0) s2t[r * CAP + p] = t;
  } else {
    const float s = rpm[t];
    float* o = outF + (size_t)t * D_DIM;
    for (int i = threadIdx.x; i < D_DIM; i += 256) o[i] = src[i] * s;
  }
}

// ---------------------------------------------------------------- WMMA GEMM
// 128x128 C tile/block, 8 waves (2x4), each 64x32 = 4x2 of 16x16x32 bf16 WMMA.
// K-tile 64, double-buffered LDS; A staged by TDM, B (fp32) converted via regs.
#define LSTRIDE 72   // ushort row stride in LDS (64 data + 8 pad) = 144B

template <int KDIM, int NDIM, bool FFN2>
__global__ void __launch_bounds__(256)
ffn_gemm(const unsigned short* __restrict__ A,
         const float* __restrict__ W,
         const float* __restrict__ bias,
         unsigned short* __restrict__ Hout,     // FFN1: bf16 [CAP,NDIM]
         const int* __restrict__ s2t,           // FFN2: slot->token
         const float* __restrict__ rpm,
         float* __restrict__ outF) {            // FFN2: [T_TOK, D_DIM]
  __shared__ __align__(16) unsigned short As[2][128 * LSTRIDE];
  __shared__ __align__(16) unsigned short Bs[2][128 * LSTRIDE];

  const int tid  = threadIdx.x;
  const int wave = tid >> 5, lane = tid & 31;
  const int wm = wave >> 2, wn = wave & 3;
  const int h2 = lane >> 4, lw = lane & 15;
  const int tM = blockIdx.x, tN = blockIdx.y;
  const unsigned short* Abase = A + (size_t)tM * 128 * KDIM;
  const float* Wbase = W + tN * 128;

  v8f acc[4][2];
#pragma unroll
  for (int mi = 0; mi < 4; ++mi)
#pragma unroll
    for (int ni = 0; ni < 2; ++ni)
#pragma unroll
      for (int r = 0; r < 8; ++r) acc[mi][ni][r] = 0.0f;

  // B staging: thread covers (k=2*band, k+1) x (4 columns); wave = one 2-row band
  const int bband0 = tid >> 5;          // 0..7, +8 per pass
  const int bn     = (tid & 31) * 4;    // column group
  float4 breg[8];
  auto loadB = [&](int k0) {
#pragma unroll
    for (int p = 0; p < 4; ++p) {
      const float* row0 = &Wbase[(size_t)(k0 + 2 * (p * 8 + bband0)) * NDIM + bn];
      breg[2 * p]     = *(const float4*)row0;
      breg[2 * p + 1] = *(const float4*)(row0 + NDIM);
    }
  };
  auto storeB = [&](int bi) {
    unsigned short* bs = &Bs[bi][0];
#pragma unroll
    for (int p = 0; p < 4; ++p) {
      const int kk = 2 * (p * 8 + bband0);
      const float4 r0 = breg[2 * p], r1 = breg[2 * p + 1];
      *(unsigned*)&bs[(bn + 0) * LSTRIDE + kk] = pack2bf(r0.x, r1.x);
      *(unsigned*)&bs[(bn + 1) * LSTRIDE + kk] = pack2bf(r0.y, r1.y);
      *(unsigned*)&bs[(bn + 2) * LSTRIDE + kk] = pack2bf(r0.z, r1.z);
      *(unsigned*)&bs[(bn + 3) * LSTRIDE + kk] = pack2bf(r0.w, r1.w);
    }
  };
  auto issueA = [&](int k0, int bi) {
    if (wave == 0)
      tdm_load_tile_a(Abase + k0, (unsigned)(size_t)&As[bi][0], KDIM);
  };

  constexpr int NS = KDIM / 64;
  issueA(0, 0);
  loadB(0);
  storeB(0);
  __builtin_amdgcn_s_wait_tensorcnt(0);
  __syncthreads();

  for (int s = 0; s < NS; ++s) {
    const int cur = s & 1, nxt = cur ^ 1;
    const bool more = (s + 1 < NS);
    if (more) { issueA((s + 1) * 64, nxt); loadB((s + 1) * 64); }

    const unsigned short* as  = &As[cur][0];
    const unsigned short* bsl = &Bs[cur][0];
#pragma unroll
    for (int ks = 0; ks < 2; ++ks) {
      FragBF a[4], b[2];
#pragma unroll
      for (int mi = 0; mi < 4; ++mi) {
        int row = wm * 64 + mi * 16 + lw;
        a[mi].q[0] = *(const uint4*)&as[row * LSTRIDE + ks * 32 + 8 * h2];
        a[mi].q[1] = *(const uint4*)&as[row * LSTRIDE + ks * 32 + 16 + 8 * h2];
      }
#pragma unroll
      for (int ni = 0; ni < 2; ++ni) {
        int cn = wn * 32 + ni * 16 + lw;
        b[ni].q[0] = *(const uint4*)&bsl[cn * LSTRIDE + ks * 32 + 8 * h2];
        b[ni].q[1] = *(const uint4*)&bsl[cn * LSTRIDE + ks * 32 + 16 + 8 * h2];
      }
#pragma unroll
      for (int mi = 0; mi < 4; ++mi)
#pragma unroll
        for (int ni = 0; ni < 2; ++ni)
          acc[mi][ni] = __builtin_amdgcn_wmma_f32_16x16x32_bf16(
              false, a[mi].v, false, b[ni].v, (short)0, acc[mi][ni], false, false);
    }
    if (more) storeB(nxt);
    __builtin_amdgcn_s_wait_tensorcnt(0);
    __syncthreads();
  }

  // epilogue
#pragma unroll
  for (int ni = 0; ni < 2; ++ni) {
    const int colBase = tN * 128 + wn * 32 + ni * 16 + lw;
    const float bv = bias[colBase];
#pragma unroll
    for (int mi = 0; mi < 4; ++mi) {
#pragma unroll
      for (int r = 0; r < 8; ++r) {
        const int row = tM * 128 + wm * 64 + mi * 16 + 8 * h2 + r;
        float v = acc[mi][ni][r] + bv;
        if (!FFN2) {
          v = fmaxf(v, 0.0f);
          Hout[(size_t)row * NDIM + colBase] = f2bf(v);
        } else {
          const int tok = s2t[row];
          if (tok >= 0)
            outF[(size_t)tok * NDIM + colBase] = v * rpm[tok];
        }
      }
    }
  }
}

// ---------------------------------------------------------------- launch
extern "C" void kernel_launch(void* const* d_in, const int* in_sizes, int n_in,
                              void* d_out, int out_size, void* d_ws, size_t ws_size,
                              hipStream_t stream) {
  const float* x   = (const float*)d_in[0];
  const float* wsw = (const float*)d_in[1];
  const float* bsw = (const float*)d_in[2];
  const float* W1  = (const float*)d_in[3];
  const float* b1  = (const float*)d_in[4];
  const float* W2  = (const float*)d_in[5];
  const float* b2  = (const float*)d_in[6];

  float* outF    = (float*)d_out;
  float* countsO = outF + (size_t)T_TOK * D_DIM;
  float* rpsO    = countsO + E_EXP;
  float* ndropO  = rpsO + E_EXP;
  float* rpmO    = ndropO + 1;

  char* p = (char*)d_ws;
  int* routes = (int*)p;                     p += (size_t)T_TOK * 4;
  int* pos    = (int*)p;                     p += (size_t)T_TOK * 4;
  int* s2t    = (int*)p;                     p += (size_t)E_EXP * CAP * 4;
  unsigned short* buf  = (unsigned short*)p; p += (size_t)E_EXP * CAP * D_DIM * 2;
  unsigned short* hbuf = (unsigned short*)p; // CAP*F_DIM*2 bytes

  init_kernel<<<(E_EXP * CAP + 255) / 256, 256, 0, stream>>>(s2t, rpsO);
  router_kernel<<<T_TOK / 8, 256, 0, stream>>>(x, wsw, bsw, routes, rpmO, rpsO);
  scan_kernel<<<1, 32, 0, stream>>>(routes, pos, countsO, ndropO);
  dispatch_kernel<<<T_TOK, 256, 0, stream>>>(x, routes, pos, rpmO, buf, s2t, outF);

  for (int e = 0; e < E_EXP; ++e) {
    ffn_gemm<D_DIM, F_DIM, false>
        <<<dim3(CAP / 128, F_DIM / 128), 256, 0, stream>>>(
            buf + (size_t)e * CAP * D_DIM,
            W1 + (size_t)e * D_DIM * F_DIM,
            b1 + (size_t)e * F_DIM,
            hbuf, nullptr, nullptr, nullptr);
    ffn_gemm<F_DIM, D_DIM, true>
        <<<dim3(CAP / 128, D_DIM / 128), 256, 0, stream>>>(
            hbuf,
            W2 + (size_t)e * F_DIM * D_DIM,
            b2 + (size_t)e * D_DIM,
            nullptr, s2t + e * CAP, rpmO, outF);
  }
}
